// DWIAngleLinear_13580686590551
// MI455X (gfx1250) — matrix operand order repaired and checked by reading, not compiled, not run on hardware
//
#include <hip/hip_runtime.h>
#include <hip/hip_bf16.h>
#include <math.h>

#define BS    256
#define IN_F  512
#define OUT_F 85742

// lamb = max(5, 1000*(1.12)^-1) = 892.857142857...
__device__ __constant__ const float INV_ONE_PLUS_LAMB = (float)(1.0 / (1.0 + 1000.0 / 1.12));
#define PI_APPROX 3.14159265f
#define EPSN 1e-12f

typedef __attribute__((ext_vector_type(16))) __bf16          v16bf;
typedef __attribute__((ext_vector_type(8)))  float           v8f;
typedef __attribute__((ext_vector_type(4)))  unsigned short  u16x4;
typedef __attribute__((ext_vector_type(8)))  unsigned short  u16x8;
typedef __attribute__((ext_vector_type(16))) unsigned short  u16x16;

static __device__ __forceinline__ unsigned short f2bf(float f) {
    // round-to-nearest-even f32 -> bf16 (bit trick, pure VALU)
    unsigned int u = __float_as_uint(f);
    u = u + 0x7FFFu + ((u >> 16) & 1u);
    return (unsigned short)(u >> 16);
}

static __device__ __forceinline__ v16bf as_v16bf(u16x16 u) {
    union { u16x16 u; v16bf b; } c;
    c.u = u;
    return c.b;
}

// ---------------------------------------------------------------------------
// Kernel 1: normalize input rows. One wave (32 lanes) per row, 8 rows/block.
// Outputs: A_bf16 [256,512] (WMMA A operand), feat_f32 [256,512], norms [256].
// ---------------------------------------------------------------------------
__global__ __launch_bounds__(256) void feat_kernel(const float* __restrict__ inp,
                                                   unsigned short* __restrict__ Abf,
                                                   float* __restrict__ feat,
                                                   float* __restrict__ norms) {
    int wave = threadIdx.x >> 5;
    int lane = threadIdx.x & 31;
    int row  = blockIdx.x * 8 + wave;            // grid = 32 -> row in [0,256)

    const float* src = inp + (size_t)row * IN_F;
    float v[16];
    float ss = 0.f;
#pragma unroll
    for (int i = 0; i < 4; ++i) {
        float4 x = *(const float4*)(src + i * 128 + lane * 4);
        v[i * 4 + 0] = x.x; v[i * 4 + 1] = x.y; v[i * 4 + 2] = x.z; v[i * 4 + 3] = x.w;
        ss += x.x * x.x + x.y * x.y + x.z * x.z + x.w * x.w;
    }
#pragma unroll
    for (int off = 16; off > 0; off >>= 1) ss += __shfl_xor(ss, off, 32);

    float nrm = sqrtf(ss);
    float inv = 1.0f / fmaxf(nrm, EPSN);

    float*          fd = feat + (size_t)row * IN_F;
    unsigned short* ad = Abf  + (size_t)row * IN_F;
#pragma unroll
    for (int i = 0; i < 4; ++i) {
        float4 o;
        o.x = v[i * 4 + 0] * inv; o.y = v[i * 4 + 1] * inv;
        o.z = v[i * 4 + 2] * inv; o.w = v[i * 4 + 3] * inv;
        *(float4*)(fd + i * 128 + lane * 4) = o;
        u16x4 b;
        b[0] = f2bf(o.x); b[1] = f2bf(o.y); b[2] = f2bf(o.z); b[3] = f2bf(o.w);
        *(u16x4*)(ad + i * 128 + lane * 4) = b;
    }
    if (lane == 0) norms[row] = nrm;
}

// ---------------------------------------------------------------------------
// Kernel 2: fused weight-normalize + copy(new_weight=weight) + WMMA GEMM +
// margin epilogue. One block per 16-column N-tile; 8 waves, each wave owns
// two 16-row M-tiles (8 waves * 32 rows = 256 = BS).
// B tile is normalized+converted to bf16 ONCE per block into LDS, stored in
// per-kstep per-lane fragment order so the inner loop is a single contiguous
// 32-byte LDS read per lane.
// ---------------------------------------------------------------------------
__global__ __launch_bounds__(256) void fused_gemm(const unsigned short* __restrict__ Abf,
                                                  const float* __restrict__ W,
                                                  const float* __restrict__ fnorm,
                                                  const int*   __restrict__ label,
                                                  float* __restrict__ out,
                                                  float* __restrict__ newW) {
    __shared__ float          s_red[256];
    __shared__ float          s_inv16[16];
    __shared__ int            s_label[BS];
    __shared__ float          s_nf[BS];
    __shared__ float          s_wf[16 * IN_F];            // 32 KB f32 weight tile
    __shared__ unsigned short s_bfrag[16 * 32 * 16];      // 16 KB bf16 B fragments

    const int t      = threadIdx.x;
    const int n_base = blockIdx.x * 16;

    s_label[t] = label[t];
    s_nf[t]    = fnorm[t];

    // ---- pass 1: read 16 weight rows, accumulate row sum-of-squares,
    //      copy rows to new_weight (counts==0 default), park f32 tile in LDS.
    {
        int rloc = t >> 4;                 // 0..15 : local row
        int sub  = t & 15;                 // 0..15 : 16 threads per row
        int n    = n_base + rloc;
        int nc   = (n < OUT_F) ? n : (OUT_F - 1);
        const float* wr = W    + (size_t)nc * IN_F;
        float*       dw = newW + (size_t)n  * IN_F;
        float ss = 0.f;
        if (n < OUT_F) {
#pragma unroll 4
            for (int j = 0; j < 32; ++j) {
                float v = wr[sub + j * 16];
                ss += v * v;
                s_wf[rloc * IN_F + sub + j * 16] = v;
                __builtin_nontemporal_store(v, dw + sub + j * 16);
            }
        } else {
#pragma unroll 4
            for (int j = 0; j < 32; ++j) {
                float v = wr[sub + j * 16];
                ss += v * v;
                s_wf[rloc * IN_F + sub + j * 16] = v;
            }
        }
        s_red[t] = ss;
    }
    __syncthreads();
    if (t < 16) {
        float ss = 0.f;
#pragma unroll
        for (int j = 0; j < 16; ++j) ss += s_red[t * 16 + j];
        s_inv16[t] = 1.0f / fmaxf(sqrtf(ss), EPSN);
    }
    __syncthreads();

    // ---- pass 2: normalize + convert B tile to bf16 in fragment order.
    // s_bfrag layout: [kstep s][lane L][16 elems]; lane L of kstep s needs
    // row (L&15), K = s*32 + (L>>4)*8 + j + (j&8)  (ISA 16-bit B layout).
    {
        int s = t >> 4;                     // kstep 0..15
#pragma unroll
        for (int l2 = 0; l2 < 2; ++l2) {
            int L   = (t & 15) * 2 + l2;    // lane 0..31
            int row = L & 15;
            int kb  = s * 32 + (L >> 4) * 8;
            float inv = s_inv16[row];
            unsigned short* dst = s_bfrag + (s * 32 + L) * 16;
            const float*    srcw = s_wf + row * IN_F + kb;
#pragma unroll
            for (int j = 0; j < 16; ++j)
                dst[j] = f2bf(srcw[j + (j & 8)] * inv);
        }
    }
    __syncthreads();

    // ---- WMMA main loop ----
    const int lane = t & 31;
    const int wave = t >> 5;
    const int l15  = lane & 15;
    const int hi   = lane >> 4;
    const int n    = n_base + l15;

    // A fragment source (bf16, pre-staged): lanes 0-15 K {0..7,16..23},
    // lanes 16-31 K {8..15,24..31} per ISA 16-bit A layout.
    const unsigned short* arow0 = Abf + (size_t)(wave * 32 + l15) * IN_F + hi * 8;
    const unsigned short* arow1 = arow0 + (size_t)16 * IN_F;
    const unsigned short* bptr  = s_bfrag + lane * 16;   // + s*512 per kstep

    v8f acc0 = {};
    v8f acc1 = {};

#pragma unroll 4
    for (int s = 0; s < 16; ++s) {
        int k = s * 32;
        v16bf b = as_v16bf(*(const u16x16*)(bptr + s * 512));   // 32B LDS read

        u16x8 a0lo = *(const u16x8*)(arow0 + k);
        u16x8 a0hi = *(const u16x8*)(arow0 + k + 16);
        u16x8 a1lo = *(const u16x8*)(arow1 + k);
        u16x8 a1hi = *(const u16x8*)(arow1 + k + 16);
        v16bf a0 = as_v16bf(__builtin_shufflevector(a0lo, a0hi,
                     0, 1, 2, 3, 4, 5, 6, 7, 8, 9, 10, 11, 12, 13, 14, 15));
        v16bf a1 = as_v16bf(__builtin_shufflevector(a1lo, a1hi,
                     0, 1, 2, 3, 4, 5, 6, 7, 8, 9, 10, 11, 12, 13, 14, 15));

        acc0 = __builtin_amdgcn_wmma_f32_16x16x32_bf16(false, a0, false, b,
                                                       (short)0, acc0, false, false);
        acc1 = __builtin_amdgcn_wmma_f32_16x16x32_bf16(false, a1, false, b,
                                                       (short)0, acc1, false, false);
    }

    // ---- epilogue: clamp, margin on one-hot positions only, scale by ||x||
    if (n < OUT_F) {
#pragma unroll
        for (int mt = 0; mt < 2; ++mt) {
            v8f acc = mt ? acc1 : acc0;
#pragma unroll
            for (int r = 0; r < 8; ++r) {
                int   m  = wave * 32 + mt * 16 + hi * 8 + r;   // C layout: VGPR r, lane half
                float c  = fminf(1.0f, fmaxf(-1.0f, acc[r]));
                float nf = s_nf[m];
                float o  = c * nf;
                if (s_label[m] == n) {
                    float c2   = c * c;
                    float cm   = 8.0f * c2 * c2 - 8.0f * c2 + 1.0f;
                    float th   = acosf(c);
                    float kk   = floorf(4.0f * th / PI_APPROX);
                    float sgn  = 1.0f - 2.0f * fmodf(kk, 2.0f);
                    float phi  = sgn * cm - 2.0f * kk;
                    o = ((phi - c) * INV_ONE_PLUS_LAMB + c) * nf;
                }
                __builtin_nontemporal_store(o, out + (size_t)m * OUT_F + n);
            }
        }
    }
}

// ---------------------------------------------------------------------------
// Kernel 3: per-class mean overwrite. One block per sample; only the first
// sample carrying each label does the work (deterministic, no atomics).
// ---------------------------------------------------------------------------
__global__ __launch_bounds__(256) void fixup_kernel(const float* __restrict__ feat,
                                                    const int*   __restrict__ label,
                                                    float* __restrict__ newW) {
    __shared__ int sl[BS];
    __shared__ int s_first, s_cnt;
    int t = threadIdx.x;
    sl[t] = label[t];
    __syncthreads();

    int s = blockIdx.x;
    int r = sl[s];
    if (t == 0) {
        int first = 1, cnt = 0;
        for (int i = 0; i < BS; ++i) {
            if (sl[i] == r) { if (i < s) first = 0; ++cnt; }
        }
        s_first = first;
        s_cnt   = cnt;
    }
    __syncthreads();
    if (!s_first) return;

    float inv = 1.0f / (float)s_cnt;
    float a0 = 0.f, a1 = 0.f;
    for (int i = 0; i < BS; ++i) {
        if (sl[i] == r) {
            a0 += feat[(size_t)i * IN_F + t];
            a1 += feat[(size_t)i * IN_F + t + 256];
        }
    }
    float* dst = newW + (size_t)r * IN_F;
    dst[t]       = a0 * inv;
    dst[t + 256] = a1 * inv;
}

// ---------------------------------------------------------------------------
extern "C" void kernel_launch(void* const* d_in, const int* in_sizes, int n_in,
                              void* d_out, int out_size, void* d_ws, size_t ws_size,
                              hipStream_t stream) {
    const float* inp    = (const float*)d_in[0];
    const float* weight = (const float*)d_in[1];
    const int*   label  = (const int*)d_in[2];

    float* out  = (float*)d_out;                         // [256, 85742]
    float* newW = out + (size_t)BS * OUT_F;              // [85742, 512]

    // workspace layout (~0.8 MB total)
    unsigned short* Abf    = (unsigned short*)d_ws;                  // 256*512 bf16
    float*          feat   = (float*)(Abf + (size_t)BS * IN_F);      // 256*512 f32
    float*          fnorms = feat + (size_t)BS * IN_F;               // 256 f32

    feat_kernel<<<BS / 8, 256, 0, stream>>>(inp, Abf, feat, fnorms);

    int ntiles = (OUT_F + 15) / 16;                      // 5359
    fused_gemm<<<ntiles, 256, 0, stream>>>(Abf, weight, fnorms, label, out, newW);

    fixup_kernel<<<BS, 256, 0, stream>>>(feat, label, newW);
}
